// KimiDeltaAttention_790273982500
// MI455X (gfx1250) — compile-verified, matrix operand order checked by reading, not run on hardware
//
#include <hip/hip_runtime.h>
#include <hip/hip_bf16.h>

typedef __attribute__((ext_vector_type(16))) __bf16 v16bf;
typedef __attribute__((ext_vector_type(8)))  __bf16 v8bf;
typedef __attribute__((ext_vector_type(8)))  float  v8f;

typedef __attribute__((ext_vector_type(4))) unsigned int tdm_u32x4;
typedef __attribute__((ext_vector_type(8))) int          tdm_i32x8;
typedef __attribute__((ext_vector_type(4))) int          tdm_i32x4;

#if defined(__AMDGCN__) && __has_builtin(__builtin_amdgcn_tensor_load_to_lds)
#define KDA_HAVE_TDM 1
#else
#define KDA_HAVE_TDM 0
#endif

// Problem dims (fixed by setup_inputs)
constexpr int Bdim = 2, Tdim = 4096, Ddim = 2048, Hh = 16, DK = 128, DV = 128, CH = 128;
constexpr int Mrows = Bdim * Tdim;   // 8192
constexpr int C2 = 2048;             // H*Dk == H*Dv == D

__device__ inline float  bf2f(__bf16 b) { return (float)b; }
__device__ inline __bf16 f2bf(float f)  { return (__bf16)f; }

__device__ inline v8f zero_v8f() {
  v8f z;
#pragma unroll
  for (int i = 0; i < 8; ++i) z[i] = 0.f;
  return z;
}

#if KDA_HAVE_TDM
// Issue a 2D TDM tile load: tile_d0 elements along stride-1 dim, tile_d1 rows,
// row stride = stride0 (all in element units). data_size code: 1=2B, 2=4B.
__device__ inline void tdm_load_2d(unsigned lds_addr, unsigned long long gaddr,
                                   int dsz_code, int tensor_d0, int tensor_d1,
                                   int tile_d0, int tile_d1, int stride0) {
  tdm_u32x4 g0;
  g0[0] = 1u;                                   // count=1, user desc, no gather
  g0[1] = lds_addr;                             // lds_addr[31:0]
  g0[2] = (unsigned)(gaddr & 0xffffffffu);      // global_addr[31:0]
  g0[3] = (unsigned)((gaddr >> 32) & 0x01ffffffu) | (2u << 30);  // addr hi | type=2
  tdm_i32x8 g1;
  g1[0] = (int)((unsigned)dsz_code << 16);      // wg_mask=0, data_size
  g1[1] = (int)(((unsigned)tensor_d0 & 0xffffu) << 16);          // dim0[15:0] @ bits63:48
  g1[2] = (int)((((unsigned)tensor_d0 >> 16) & 0xffffu) |
                (((unsigned)tensor_d1 & 0xffffu) << 16));        // dim0 hi | dim1 lo
  g1[3] = (int)((((unsigned)tensor_d1 >> 16) & 0xffffu) |
                (((unsigned)tile_d0 & 0xffffu) << 16));          // dim1 hi | tile_dim0
  g1[4] = (int)((unsigned)tile_d1 & 0xffffu);   // tile_dim1 | tile_dim2=0
  g1[5] = stride0;                              // tensor_dim0_stride[31:0]
  g1[6] = 0;                                    // stride0 hi | stride1 lo
  g1[7] = 0;
  tdm_i32x4 gz;
  gz[0] = gz[1] = gz[2] = gz[3] = 0;
#if defined(__clang_major__) && __clang_major__ >= 23
  tdm_i32x8 gz8;
#pragma unroll
  for (int i = 0; i < 8; ++i) gz8[i] = 0;
  __builtin_amdgcn_tensor_load_to_lds(g0, g1, gz, gz, gz8, 0);
#else
  __builtin_amdgcn_tensor_load_to_lds(g0, g1, gz, gz, 0);
#endif
}
#endif

// Load one 16x32 bf16 WMMA operand fragment from a row-major buffer.
// Per ISA 7.12.2: lane<16 holds K{0..7,16..23}, lane>=16 holds K{8..15,24..31}.
__device__ inline v16bf frag16(const __bf16* rowp, int kk, int sel) {
  v8bf lo = *(const v8bf*)(rowp + kk + sel * 8);
  v8bf hi = *(const v8bf*)(rowp + kk + 16 + sel * 8);
  v16bf r;
#pragma unroll
  for (int i = 0; i < 8; ++i) { r[i] = lo[i]; r[i + 8] = hi[i]; }
  return r;
}

// acc += X(128x128 row-major) @ Y^T (Y row-major, N rows of K).
// All fragments preloaded per K-step so the 8 WMMAs issue back-to-back.
__device__ inline void gemm_nt_128(const __bf16* X, const __bf16* Y, v8f acc[8],
                                   int m0, int lane) {
  int sel = lane >> 4, ll = lane & 15;
  const __bf16* xrow = X + (m0 + ll) * 128;
#pragma unroll
  for (int ks = 0; ks < 4; ++ks) {
    int kk = ks * 32;
    v16bf af = frag16(xrow, kk, sel);
    v16bf bfr[8];
#pragma unroll
    for (int n = 0; n < 8; ++n) bfr[n] = frag16(Y + (n * 16 + ll) * 128, kk, sel);
#pragma unroll
    for (int n = 0; n < 8; ++n)
      acc[n] = __builtin_amdgcn_wmma_f32_16x16x32_bf16(false, af, false, bfr[n],
                                                       (short)0, acc[n], false, false);
  }
}

// ---------------------------------------------------------------------------
// fp32 -> bf16 cast
// ---------------------------------------------------------------------------
__global__ void kda_cast_f32_bf16(const float* __restrict__ s, __bf16* __restrict__ d, int n) {
  for (int i = blockIdx.x * blockDim.x + threadIdx.x; i < n; i += gridDim.x * blockDim.x)
    d[i] = f2bf(s[i]);
}

// ---------------------------------------------------------------------------
// Generic bf16 GEMM: C[M,N] = A[M,K] @ W[K,N], row-major, f32 or bf16 output.
// Block tile 128x128, K-step 32, 256 threads (8 waves), WMMA bf16.
// A tile staged by the Tensor Data Mover when available.
// ---------------------------------------------------------------------------
__global__ void __launch_bounds__(256)
kda_gemm_bf16(const __bf16* __restrict__ A, const __bf16* __restrict__ W,
              float* __restrict__ Cf, __bf16* __restrict__ Cb,
              int Mn, int Nn, int Kn) {
  __shared__ __bf16 At[128 * 32];   // A tile, row-major [128][32]
  __shared__ __bf16 Wt[128 * 32];   // W tile transposed: [n][k]
  int tid = threadIdx.x, lane = tid & 31, wv = tid >> 5;
  int mBase = blockIdx.y * 128, nBase = blockIdx.x * 128;

  v8f acc[8];
#pragma unroll
  for (int n = 0; n < 8; ++n) acc[n] = zero_v8f();

  int wrow = tid >> 3, wnb = (tid & 7) * 16;   // W-tile loader mapping
  int sel = lane >> 4, ll = lane & 15;

  for (int k0 = 0; k0 < Kn; k0 += 32) {
    __syncthreads();
#if KDA_HAVE_TDM
    // Async tensor DMA: 128 rows x 32 cols bf16 tile of A -> LDS (wave 0 issues)
    if (wv == 0) {
      unsigned long long ga =
          (unsigned long long)(size_t)(A + (size_t)mBase * Kn + k0);
      tdm_load_2d((unsigned)(size_t)At, ga, /*dsz=*/1,
                  /*tensor_d0=*/Kn, /*tensor_d1=*/Mn,
                  /*tile_d0=*/32, /*tile_d1=*/128, /*stride0=*/Kn);
    }
#else
    {
      int arow = tid >> 1, acb = (tid & 1) * 16;
      const __bf16* ap = A + (size_t)(mBase + arow) * Kn + k0 + acb;
      *(v8bf*)(At + arow * 32 + acb)     = *(const v8bf*)ap;
      *(v8bf*)(At + arow * 32 + acb + 8) = *(const v8bf*)(ap + 8);
    }
#endif
    // Stage W tile transposed (manual: TDM cannot transpose)
    __bf16 wl[16];
    if (nBase + wnb + 16 <= Nn) {
      const __bf16* wp = W + (size_t)(k0 + wrow) * Nn + nBase + wnb;
      __builtin_prefetch(wp + (size_t)32 * Nn, 0, 1);
      v8bf a0 = *(const v8bf*)wp, a1 = *(const v8bf*)(wp + 8);
#pragma unroll
      for (int j = 0; j < 8; ++j) { wl[j] = a0[j]; wl[8 + j] = a1[j]; }
    } else {
#pragma unroll
      for (int j = 0; j < 16; ++j) {
        int col = nBase + wnb + j;
        wl[j] = (col < Nn) ? W[(size_t)(k0 + wrow) * Nn + col] : f2bf(0.f);
      }
    }
#pragma unroll
    for (int j = 0; j < 16; ++j) Wt[(wnb + j) * 32 + wrow] = wl[j];
#if KDA_HAVE_TDM
    if (wv == 0) __builtin_amdgcn_s_wait_tensorcnt(0);
#endif
    __syncthreads();

    v16bf af = frag16(At + (wv * 16 + ll) * 32, 0, sel);
    v16bf bfr[8];
#pragma unroll
    for (int n = 0; n < 8; ++n) bfr[n] = frag16(Wt + (n * 16 + ll) * 32, 0, sel);
#pragma unroll
    for (int n = 0; n < 8; ++n)
      acc[n] = __builtin_amdgcn_wmma_f32_16x16x32_bf16(false, af, false, bfr[n],
                                                       (short)0, acc[n], false, false);
  }

  // Epilogue: C/D layout — lane l, vgpr r -> row = r + 8*(l>>4), col = l&15
  int rbase = mBase + wv * 16 + (lane >> 4) * 8;
  int cl = lane & 15;
#pragma unroll
  for (int n = 0; n < 8; ++n) {
    int col = nBase + n * 16 + cl;
    if (col < Nn) {
#pragma unroll
      for (int r = 0; r < 8; ++r) {
        size_t idx = (size_t)(rbase + r) * Nn + col;
        if (Cf) Cf[idx] = acc[n][r];
        if (Cb) Cb[idx] = f2bf(acc[n][r]);
      }
    }
  }
}

// ---------------------------------------------------------------------------
// Causal depthwise conv (K=4) + SiLU (+ optional per-head L2 norm)
// mode: 0 = v (silu only), 1 = k (silu + l2norm), 2 = q (silu + l2norm * Dk^-0.5)
// ---------------------------------------------------------------------------
__global__ void __launch_bounds__(256)
kda_conv_silu(const float* __restrict__ X, const float* __restrict__ w,
              __bf16* __restrict__ out, int mode) {
  int bt = blockIdx.x;
  int b = bt / Tdim, t = bt % Tdim;
  int c0 = threadIdx.x * 8;
  float y[8];
#pragma unroll
  for (int j = 0; j < 8; ++j) y[j] = 0.f;
#pragma unroll
  for (int i = 0; i < 4; ++i) {
    int tt = t - 3 + i;
    if (tt >= 0) {
      const float* xp = X + (size_t)(b * Tdim + tt) * C2 + c0;
#pragma unroll
      for (int j = 0; j < 8; ++j) y[j] += xp[j] * w[(c0 + j) * 4 + i];
    }
  }
  float ss = 0.f;
#pragma unroll
  for (int j = 0; j < 8; ++j) {
    float v = y[j];
    v = v / (1.f + expf(-v));     // silu
    y[j] = v;
    ss += v * v;
  }
  float scale = 1.f;
  if (mode > 0) {
    ss += __shfl_xor(ss, 1, 32);
    ss += __shfl_xor(ss, 2, 32);
    ss += __shfl_xor(ss, 4, 32);
    ss += __shfl_xor(ss, 8, 32);
    scale = rsqrtf(ss + 1e-6f);
    if (mode == 2) scale *= 0.08838834764831845f;  // Dk^-0.5
  }
  __bf16* op = out + (size_t)bt * C2 + c0;
#pragma unroll
  for (int j = 0; j < 8; ++j) op[j] = f2bf(y[j] * scale);
}

// ---------------------------------------------------------------------------
__global__ void kda_gk(float* __restrict__ g, const float* __restrict__ A_log,
                       const float* __restrict__ dt_bias, int n) {
  for (int i = blockIdx.x * blockDim.x + threadIdx.x; i < n; i += gridDim.x * blockDim.x) {
    int c = i & (C2 - 1);
    int h = c >> 7;
    float x = g[i] + dt_bias[c];
    float sp = (x > 20.f) ? x : log1pf(expf(x));
    g[i] = -expf(A_log[h]) * sp;
  }
}

__global__ void kda_sigmoid(float* __restrict__ x, int n) {
  for (int i = blockIdx.x * blockDim.x + threadIdx.x; i < n; i += gridDim.x * blockDim.x)
    x[i] = 1.f / (1.f + expf(-x[i]));
}

// ---------------------------------------------------------------------------
// Chunked gated delta-rule scan. One block per (b,h); 256 threads (8 waves).
// ---------------------------------------------------------------------------
constexpr size_t SC_S    = 0;            // f32 [128][128]
constexpr size_t SC_G    = 65536;        // f32 [128][128]  (gk tile -> cumsum -> A scores)
constexpr size_t SC_QG   = 131072;       // bf16 [t][dk]
constexpr size_t SC_KD   = 163840;       // bf16 [t][dk]    (later SbT [dv][dk])
constexpr size_t SC_KGB  = 196608;       // bf16 [t][dk]    (later P [t][s])
constexpr size_t SC_KUT  = 229376;       // bf16 [dk][t]
constexpr size_t SC_UT   = 262144;       // bf16 [dv][t]
constexpr size_t SC_BETA = 294912;       // f32 [128]
constexpr size_t SC_GTOT = 295424;       // f32 [128]
constexpr size_t SC_EGT  = 295936;       // f32 [128]
constexpr size_t SCAN_SMEM = 296448;

__global__ void __launch_bounds__(256)
kda_scan(const __bf16* __restrict__ Q, const __bf16* __restrict__ K,
         const __bf16* __restrict__ V, const float* __restrict__ GK,
         const float* __restrict__ BETA, float* __restrict__ O) {
  extern __shared__ char smem[];
  float*  S    = (float*)(smem + SC_S);
  float*  Gc   = (float*)(smem + SC_G);
  __bf16* sQg  = (__bf16*)(smem + SC_QG);
  __bf16* sKd  = (__bf16*)(smem + SC_KD);
  __bf16* sKgb = (__bf16*)(smem + SC_KGB);
  __bf16* sKu  = (__bf16*)(smem + SC_KUT);
  __bf16* sUT  = (__bf16*)(smem + SC_UT);
  float*  betaS = (float*)(smem + SC_BETA);
  float*  gtot  = (float*)(smem + SC_GTOT);
  float*  egtot = (float*)(smem + SC_EGT);

  int tid = threadIdx.x, lane = tid & 31, wv = tid >> 5;
  int bh = blockIdx.x;
  int b = bh / Hh, h = bh % Hh;
  size_t headOff = (size_t)h * DK;

  for (int i = tid; i < DK * DV; i += 256) S[i] = 0.f;
  __syncthreads();

  for (int ci = 0; ci < Tdim / CH; ++ci) {
    int t0 = ci * CH;

    // (a) bring gk chunk tile [128 t][128 dk] into LDS, then in-place cumsum
#if KDA_HAVE_TDM
    if (wv == 0) {
      unsigned long long ga = (unsigned long long)(size_t)(
          GK + (size_t)(b * Tdim + t0) * C2 + headOff);
      tdm_load_2d((unsigned)(size_t)Gc, ga, /*dsz=*/2,
                  /*tensor_d0=*/C2, /*tensor_d1=*/Bdim * Tdim,
                  /*tile_d0=*/DK, /*tile_d1=*/CH, /*stride0=*/C2);
      __builtin_amdgcn_s_wait_tensorcnt(0);
    }
    __syncthreads();
    if (tid < DK) {
      int dk = tid;
      float acc = 0.f;
      for (int t = 0; t < CH; ++t) {
        acc += Gc[t * DK + dk];
        Gc[t * DK + dk] = acc;
      }
      gtot[dk] = acc;
      egtot[dk] = expf(acc);
      betaS[dk] = BETA[(size_t)(b * Tdim + t0 + dk) * Hh + h];  // dk doubles as t index
    }
#else
    if (tid < DK) {
      int dk = tid;
      float acc = 0.f;
      for (int t = 0; t < CH; ++t) {
        acc += GK[(size_t)(b * Tdim + t0 + t) * C2 + headOff + dk];
        Gc[t * DK + dk] = acc;
      }
      gtot[dk] = acc;
      egtot[dk] = expf(acc);
      betaS[dk] = BETA[(size_t)(b * Tdim + t0 + dk) * Hh + h];
    }
#endif
    __syncthreads();

    // (c) stage decayed bf16 operand matrices
    for (int i = tid; i < CH * DK; i += 256) {
      int t = i >> 7, dk = i & 127;
      size_t gidx = (size_t)(b * Tdim + t0 + t) * C2 + headOff + dk;
      float qv = bf2f(Q[gidx]), kv = bf2f(K[gidx]);
      float g = Gc[i];
      float eg = expf(g);
      sQg[i]  = f2bf(qv * eg);
      sKd[i]  = f2bf(kv * expf(-g));
      sKgb[i] = f2bf(kv * eg * betaS[t]);
      sKu[dk * CH + t] = f2bf(kv * expf(gtot[dk] - g));
    }
    __syncthreads();

    // (e) A = strict_lower(Kgb @ Kd^T) -> Gc (f32)
    {
      v8f acc[8];
#pragma unroll
      for (int n = 0; n < 8; ++n) acc[n] = zero_v8f();
      gemm_nt_128(sKgb, sKd, acc, wv * 16, lane);
      int rb = wv * 16 + (lane >> 4) * 8, cl = lane & 15;
#pragma unroll
      for (int n = 0; n < 8; ++n) {
        int col = n * 16 + cl;
#pragma unroll
        for (int r = 0; r < 8; ++r) {
          int row = rb + r;
          Gc[row * DK + col] = (col < row) ? acc[n][r] : 0.f;
        }
      }
    }
    __syncthreads();

    // (g) forward substitution: each of 128 lanes owns one dv column
    if (tid < DV) {
      int dv = tid;
      for (int t = 0; t < CH; ++t) {
        float a = betaS[t] * bf2f(V[(size_t)(b * Tdim + t0 + t) * C2 + headOff + dv]);
        const float* Arow = Gc + t * DK;
        for (int s = 0; s < t; ++s) a -= Arow[s] * bf2f(sUT[dv * CH + s]);
        sUT[dv * CH + t] = f2bf(a);
      }
    }
    __syncthreads();

    // (i) P = mask_{s<=t}(Qg @ Kd^T) -> sKgb (Kgb dead)
    {
      v8f acc[8];
#pragma unroll
      for (int n = 0; n < 8; ++n) acc[n] = zero_v8f();
      gemm_nt_128(sQg, sKd, acc, wv * 16, lane);
      int rb = wv * 16 + (lane >> 4) * 8, cl = lane & 15;
#pragma unroll
      for (int n = 0; n < 8; ++n) {
        int col = n * 16 + cl;
#pragma unroll
        for (int r = 0; r < 8; ++r) {
          int row = rb + r;
          sKgb[row * CH + col] = f2bf((col <= row) ? acc[n][r] : 0.f);
        }
      }
    }
    __syncthreads();

    // (j) SbT[dv][dk] = bf16(S[dk][dv]) -> sKd (Kd dead)
    for (int i = tid; i < DK * DV; i += 256) {
      int dv = i >> 7, dk = i & 127;
      sKd[i] = f2bf(S[dk * DV + dv]);
    }
    __syncthreads();

    // (l) o = Qg @ SbT^T + P @ UT^T  -> global (f32)
    {
      v8f acc[8];
#pragma unroll
      for (int n = 0; n < 8; ++n) acc[n] = zero_v8f();
      gemm_nt_128(sQg, sKd, acc, wv * 16, lane);   // inter-chunk
      gemm_nt_128(sKgb, sUT, acc, wv * 16, lane);  // intra-chunk
      int rb = wv * 16 + (lane >> 4) * 8, cl = lane & 15;
#pragma unroll
      for (int n = 0; n < 8; ++n) {
        int col = n * 16 + cl;
#pragma unroll
        for (int r = 0; r < 8; ++r) {
          int row = rb + r;
          O[(size_t)(b * Tdim + t0 + row) * C2 + headOff + col] = acc[n][r];
        }
      }
    }
    __syncthreads();

    // (n) S = diag(e^Gtot) S + Ku^T @ U
    {
      v8f acc[8];
      int rb = wv * 16 + (lane >> 4) * 8, cl = lane & 15;
#pragma unroll
      for (int n = 0; n < 8; ++n) {
        int col = n * 16 + cl;
#pragma unroll
        for (int r = 0; r < 8; ++r) {
          int row = rb + r;
          acc[n][r] = egtot[row] * S[row * DV + col];
        }
      }
      gemm_nt_128(sKu, sUT, acc, wv * 16, lane);
#pragma unroll
      for (int n = 0; n < 8; ++n) {
        int col = n * 16 + cl;
#pragma unroll
        for (int r = 0; r < 8; ++r) {
          int row = rb + r;
          S[row * DV + col] = acc[n][r];
        }
      }
    }
    __syncthreads();
  }
}

// ---------------------------------------------------------------------------
// FusedRMSNormGated (sigmoid): per-head RMS over Dv, * o_norm_w, * sigmoid(gate)
// ---------------------------------------------------------------------------
__global__ void __launch_bounds__(256)
kda_postnorm(const float* __restrict__ O, const float* __restrict__ GATE,
             const float* __restrict__ g_b2, const float* __restrict__ o_norm_w,
             __bf16* __restrict__ out) {
  int bt = blockIdx.x;
  int c0 = threadIdx.x * 8;
  const float* op = O + (size_t)bt * C2 + c0;
  float y[8];
  float ss = 0.f;
#pragma unroll
  for (int j = 0; j < 8; ++j) { y[j] = op[j]; ss += y[j] * y[j]; }
  ss += __shfl_xor(ss, 1, 32);
  ss += __shfl_xor(ss, 2, 32);
  ss += __shfl_xor(ss, 4, 32);
  ss += __shfl_xor(ss, 8, 32);
  float r = rsqrtf(ss * (1.f / 128.f) + 1e-5f);
#pragma unroll
  for (int j = 0; j < 8; ++j) {
    int c = c0 + j;
    int dv = c & 127;
    float gate = GATE[(size_t)bt * C2 + c] + g_b2[c];
    float val = y[j] * r * o_norm_w[dv] * (1.f / (1.f + expf(-gate)));
    out[(size_t)bt * C2 + c] = f2bf(val);
  }
}

// ---------------------------------------------------------------------------
extern "C" void kernel_launch(void* const* d_in, const int* in_sizes, int n_in,
                              void* d_out, int out_size, void* d_ws, size_t ws_size,
                              hipStream_t stream) {
  const float* hs       = (const float*)d_in[0];
  const float* q_w      = (const float*)d_in[1];
  const float* k_w      = (const float*)d_in[2];
  const float* v_w      = (const float*)d_in[3];
  const float* q_conv_w = (const float*)d_in[4];
  const float* k_conv_w = (const float*)d_in[5];
  const float* v_conv_w = (const float*)d_in[6];
  const float* f_w1     = (const float*)d_in[7];
  const float* f_w2     = (const float*)d_in[8];
  const float* b_w      = (const float*)d_in[9];
  const float* A_log    = (const float*)d_in[10];
  const float* dt_bias  = (const float*)d_in[11];
  const float* g_w1     = (const float*)d_in[12];
  const float* g_w2     = (const float*)d_in[13];
  const float* g_b2     = (const float*)d_in[14];
  const float* o_norm_w = (const float*)d_in[15];
  const float* o_w      = (const float*)d_in[16];

  char* base = (char*)d_ws;
  size_t off = 0;
  auto alloc = [&](size_t elems, size_t esz) -> void* {
    void* p = base + off;
    off += (elems * esz + 255) & ~(size_t)255;
    return p;
  };

  __bf16* hs_bf   = (__bf16*)alloc((size_t)Mrows * Ddim, 2);
  __bf16* qw_bf   = (__bf16*)alloc((size_t)Ddim * C2, 2);
  __bf16* kw_bf   = (__bf16*)alloc((size_t)Ddim * C2, 2);
  __bf16* vw_bf   = (__bf16*)alloc((size_t)Ddim * C2, 2);
  __bf16* ow_bf   = (__bf16*)alloc((size_t)C2 * Ddim, 2);
  __bf16* fw1_bf  = (__bf16*)alloc((size_t)Ddim * 128, 2);
  __bf16* fw2_bf  = (__bf16*)alloc((size_t)128 * C2, 2);
  __bf16* gw1_bf  = (__bf16*)alloc((size_t)Ddim * 128, 2);
  __bf16* gw2_bf  = (__bf16*)alloc((size_t)128 * C2, 2);
  __bf16* bw_bf   = (__bf16*)alloc((size_t)Ddim * Hh, 2);
  float*  q_lin   = (float*)alloc((size_t)Mrows * C2, 4);
  float*  k_lin   = (float*)alloc((size_t)Mrows * C2, 4);
  float*  v_lin   = (float*)alloc((size_t)Mrows * C2, 4);
  __bf16* g1_bf   = (__bf16*)alloc((size_t)Mrows * 128, 2);
  __bf16* gate1_bf= (__bf16*)alloc((size_t)Mrows * 128, 2);
  float*  g_raw   = (float*)alloc((size_t)Mrows * C2, 4);   // becomes gk in place
  float*  gate_raw= (float*)alloc((size_t)Mrows * C2, 4);
  float*  beta_raw= (float*)alloc((size_t)Mrows * Hh, 4);   // becomes beta in place
  __bf16* q_bf    = (__bf16*)alloc((size_t)Mrows * C2, 2);
  __bf16* k_bf    = (__bf16*)alloc((size_t)Mrows * C2, 2);
  __bf16* v_bf    = (__bf16*)alloc((size_t)Mrows * C2, 2);
  float*  o_f32   = (float*)alloc((size_t)Mrows * C2, 4);
  __bf16* o_bf    = (__bf16*)alloc((size_t)Mrows * C2, 2);

  auto cast = [&](const float* s, __bf16* d, int n) {
    int blocks = (n + 256 * 8 - 1) / (256 * 8);
    kda_cast_f32_bf16<<<blocks, 256, 0, stream>>>(s, d, n);
  };
  cast(hs,   hs_bf,  Mrows * Ddim);
  cast(q_w,  qw_bf,  Ddim * C2);
  cast(k_w,  kw_bf,  Ddim * C2);
  cast(v_w,  vw_bf,  Ddim * C2);
  cast(o_w,  ow_bf,  C2 * Ddim);
  cast(f_w1, fw1_bf, Ddim * 128);
  cast(f_w2, fw2_bf, 128 * C2);
  cast(g_w1, gw1_bf, Ddim * 128);
  cast(g_w2, gw2_bf, 128 * C2);
  cast(b_w,  bw_bf,  Ddim * Hh);

  auto gemm = [&](const __bf16* A, const __bf16* W, float* Cf, __bf16* Cb,
                  int M_, int N_, int K_) {
    dim3 g((N_ + 127) / 128, (M_ + 127) / 128);
    kda_gemm_bf16<<<g, 256, 0, stream>>>(A, W, Cf, Cb, M_, N_, K_);
  };

  // projections
  gemm(hs_bf, qw_bf, q_lin, nullptr, Mrows, C2, Ddim);
  gemm(hs_bf, kw_bf, k_lin, nullptr, Mrows, C2, Ddim);
  gemm(hs_bf, vw_bf, v_lin, nullptr, Mrows, C2, Ddim);
  // decay-gate chain
  gemm(hs_bf, fw1_bf, nullptr, g1_bf, Mrows, 128, Ddim);
  gemm(g1_bf, fw2_bf, g_raw, nullptr, Mrows, C2, 128);
  // output-gate chain
  gemm(hs_bf, gw1_bf, nullptr, gate1_bf, Mrows, 128, Ddim);
  gemm(gate1_bf, gw2_bf, gate_raw, nullptr, Mrows, C2, 128);
  // beta
  gemm(hs_bf, bw_bf, beta_raw, nullptr, Mrows, Hh, Ddim);

  // conv + silu (+ l2norm for q/k)
  kda_conv_silu<<<Mrows, 256, 0, stream>>>(q_lin, q_conv_w, q_bf, 2);
  kda_conv_silu<<<Mrows, 256, 0, stream>>>(k_lin, k_conv_w, k_bf, 1);
  kda_conv_silu<<<Mrows, 256, 0, stream>>>(v_lin, v_conv_w, v_bf, 0);

  kda_gk<<<4096, 256, 0, stream>>>(g_raw, A_log, dt_bias, Mrows * C2);
  kda_sigmoid<<<256, 256, 0, stream>>>(beta_raw, Mrows * Hh);

  // chunked gated delta-rule scan (1 block per (b,h))
  kda_scan<<<Bdim * Hh, 256, SCAN_SMEM, stream>>>(q_bf, k_bf, v_bf, g_raw, beta_raw, o_f32);

  kda_postnorm<<<Mrows, 256, 0, stream>>>(o_f32, gate_raw, g_b2, o_norm_w, o_bf);

  // final projection -> d_out (f32)
  gemm(o_bf, ow_bf, (float*)d_out, nullptr, Mrows, Ddim, C2);
}
